// JKNet_24498493456721
// MI455X (gfx1250) — matrix operand order repaired and checked by reading, not compiled, not run on hardware
//
#include <hip/hip_runtime.h>

// ---------------------------------------------------------------------------
// JKNet (6-layer GCN + JumpingKnowledge-max + FC + log_softmax) for MI455X
//
// N=100000 nodes, E=1600000 edges, F_IN=128, H=64, C=40, L=6
//
// Strategy:
//  * GEMMs (h@W) in f16 via v_wmma_f32_16x16x32_f16 (fp32 accumulate).
//  * Weights pre-packed into CDNA5 B-fragment layout once per launch, then
//    staged per-block into LDS with global_load_async_to_lds_b128 (ASYNCcnt)
//    and consumed via ds_load_b128 -> VMEM pipe is dedicated to streaming A.
//  * GCN aggregation: fp32 gather + global_atomic_add_f32 (L2-resident:
//    whole working set ~90MB < 192MB L2).
//  * Self-loop contribution folded into the agg-init kernel (agg = T*dis^2).
//  * Bias+ReLU+JK-max+f16-convert fused in one epilogue kernel.
//  * FC + log_softmax fully fp32, LDS tiled.
// ---------------------------------------------------------------------------

#define N_NODES 100000
#define N_EDGES 1600000
#define F_IN    128
#define HDIM    64
#define NCLS    40
#define NLAYERS 6

typedef __attribute__((ext_vector_type(16))) _Float16 v16h;
typedef __attribute__((ext_vector_type(8)))  _Float16 v8h;
typedef __attribute__((ext_vector_type(8)))  float    v8f;

// ---------------------------------------------------------------------------
// Degree / normalization
// ---------------------------------------------------------------------------
__global__ void deg_init_kernel(float* __restrict__ deg, int n) {
    int i = blockIdx.x * blockDim.x + threadIdx.x;
    if (i < n) deg[i] = 1.0f;                       // self-loop contributes 1
}

__global__ void deg_count_kernel(const int* __restrict__ dst,
                                 float* __restrict__ deg, int e) {
    int i = blockIdx.x * blockDim.x + threadIdx.x;
    if (i >= e) return;
    __hip_atomic_fetch_add(&deg[dst[i]], 1.0f,
                           __ATOMIC_RELAXED, __HIP_MEMORY_SCOPE_AGENT);
}

__global__ void deg_rsqrt_kernel(float* __restrict__ dis, int n) {
    int i = blockIdx.x * blockDim.x + threadIdx.x;
    if (i >= n) return;
    float d = dis[i];
    dis[i] = d > 0.0f ? rsqrtf(d) : 0.0f;
}

__global__ void edge_norm_kernel(const int* __restrict__ src,
                                 const int* __restrict__ dst,
                                 const float* __restrict__ dis,
                                 float* __restrict__ norm, int e) {
    int i = blockIdx.x * blockDim.x + threadIdx.x;
    if (i >= e) return;
    norm[i] = dis[src[i]] * dis[dst[i]];
}

// ---------------------------------------------------------------------------
// fp32 -> f16 convert (activations feeding WMMA)
// ---------------------------------------------------------------------------
__global__ void cvt_f16_kernel(const float* __restrict__ in,
                               _Float16* __restrict__ out, int total) {
    int i = blockIdx.x * blockDim.x + threadIdx.x;
    if (i < total) out[i] = (_Float16)in[i];
}

// ---------------------------------------------------------------------------
// Pack weight W[K,64] (row-major fp32) into WMMA B-fragment order (f16).
// Layout: [kt][nt][lane][e], e in 0..15.
//   lane 0-15  hold column n = nt*16 + lane,    K = kt*32 + e (e<8), kt*32 + 16 + (e-8) (e>=8)
//   lane 16-31 hold column n = nt*16 + lane-16, K = kt*32 + 8 + e (e<8), kt*32 + 24 + (e-8) (e>=8)
// (VGPR j packs K=2j in [15:0], K=2j+1 in [31:16] -> flat element index e.)
// ---------------------------------------------------------------------------
__global__ void pack_w_kernel(const float* __restrict__ W,
                              _Float16* __restrict__ Bp, int K) {
    int idx = blockIdx.x * blockDim.x + threadIdx.x;
    if (idx >= K * 64) return;
    int e    = idx & 15;
    int lane = (idx >> 4) & 31;
    int nt   = (idx >> 9) & 3;
    int kt   = idx >> 11;
    int n       = nt * 16 + (lane & 15);
    int k_local = ((lane >> 4) << 3) + (e & 7) + ((e >> 3) << 4);
    int k       = kt * 32 + k_local;
    Bp[idx] = (_Float16)W[k * 64 + n];
}

// ---------------------------------------------------------------------------
// GEMM: T[M,64] = A[M,K] @ W[K,64], A f16 row-major, W pre-packed fragments.
// One wave computes a 16x64 output tile (4 accumulators), K in steps of 32.
// 256 threads = 8 waves = 128 rows per block. M % 16 == 0 (100000 = 6250*16).
// Packed W staged block-wide into LDS via async copy; WMMA feeds off DS.
// ---------------------------------------------------------------------------
__global__ void gemm_wmma_kernel(const _Float16* __restrict__ A,
                                 const _Float16* __restrict__ Bp,
                                 float* __restrict__ T,
                                 int M, int K) {
    __shared__ _Float16 sB[4 * 4 * 32 * 16];        // up to K=128: 16 KB

    // ---- stage packed W into LDS (async copy path, uniform trip count) -----
    {
        const int nBytes = K * 64 * 2;              // 8 KB (K=64) / 16 KB (K=128)
        const unsigned ldsBase = (unsigned)(uintptr_t)(&sB[0]);
        const char* gBase = (const char*)Bp;
        for (int ofs = threadIdx.x * 16; ofs < nBytes; ofs += 256 * 16) {
            unsigned ldsAddr = ldsBase + (unsigned)ofs;
            const char* g = gBase + ofs;
            asm volatile("global_load_async_to_lds_b128 %0, %1, off"
                         :: "v"(ldsAddr), "v"(g) : "memory");
        }
        asm volatile("s_wait_asynccnt 0x0" ::: "memory");
    }
    __syncthreads();

    const int lane    = threadIdx.x & 31;
    const int wave    = threadIdx.x >> 5;
    const int rowTile = blockIdx.x * 8 + wave;
    const int nTiles  = M >> 4;
    if (rowTile >= nTiles) return;                  // wave-uniform: EXEC all-1s

    const int m     = rowTile * 16 + (lane & 15);   // A row for this lane
    const int kHalf = (lane >> 4) * 8;              // K sub-offset per ISA layout

    v8f acc0 = {}, acc1 = {}, acc2 = {}, acc3 = {};
    const int nkt = K >> 5;

    for (int kt = 0; kt < nkt; ++kt) {
        // A fragment: lane holds K = kHalf + [0..7] and kHalf + 16 + [0..7]
        const _Float16* arow = A + (size_t)m * K + kt * 32 + kHalf;
        if (kt + 1 < nkt)
            __builtin_prefetch(arow + 32, 0, 3);    // global_prefetch_b8 (near)
        v8h a_lo = *(const v8h*)(arow);
        v8h a_hi = *(const v8h*)(arow + 16);
        v16h a;
#pragma unroll
        for (int i = 0; i < 8; ++i) { a[i] = a_lo[i]; a[8 + i] = a_hi[i]; }

        // B fragments from LDS: 512 halfs per (kt,nt) tile
        const _Float16* bbase = sB + ((size_t)kt * 4 * 32 + lane) * 16;
        v16h b0 = *(const v16h*)(bbase + 0 * 512);
        v16h b1 = *(const v16h*)(bbase + 1 * 512);
        v16h b2 = *(const v16h*)(bbase + 2 * 512);
        v16h b3 = *(const v16h*)(bbase + 3 * 512);

        acc0 = __builtin_amdgcn_wmma_f32_16x16x32_f16(false, a, false, b0,
                                                      (short)0, acc0, false, false);
        acc1 = __builtin_amdgcn_wmma_f32_16x16x32_f16(false, a, false, b1,
                                                      (short)0, acc1, false, false);
        acc2 = __builtin_amdgcn_wmma_f32_16x16x32_f16(false, a, false, b2,
                                                      (short)0, acc2, false, false);
        acc3 = __builtin_amdgcn_wmma_f32_16x16x32_f16(false, a, false, b3,
                                                      (short)0, acc3, false, false);
    }

    // D layout: VGPR v, lane l -> row = v + 8*(l/16), col = l%16
    const int rbase = rowTile * 16 + ((lane >> 4) << 3);
    const int cbase = lane & 15;
    float* trow = T + (size_t)rbase * 64 + cbase;
#pragma unroll
    for (int v = 0; v < 8; ++v) {
        trow[(size_t)v * 64 +  0] = acc0[v];
        trow[(size_t)v * 64 + 16] = acc1[v];
        trow[(size_t)v * 64 + 32] = acc2[v];
        trow[(size_t)v * 64 + 48] = acc3[v];
    }
}

// ---------------------------------------------------------------------------
// agg[n][f] = T[n][f] * dis[n]^2   (self-loop term; also serves as zero-init)
// ---------------------------------------------------------------------------
__global__ void self_init_kernel(const float* __restrict__ T,
                                 const float* __restrict__ dis,
                                 float* __restrict__ agg, int total) {
    int i = blockIdx.x * blockDim.x + threadIdx.x;
    if (i >= total) return;
    float dn = dis[i >> 6];
    agg[i] = T[i] * dn * dn;
}

// ---------------------------------------------------------------------------
// Edge scatter: agg[dst] += T[src] * norm.  One lane per (edge, feature).
// All traffic stays in L2 (T/agg are 25.6MB each << 192MB).
// ---------------------------------------------------------------------------
__global__ void edge_scatter_kernel(const int* __restrict__ src,
                                    const int* __restrict__ dst,
                                    const float* __restrict__ norm,
                                    const float* __restrict__ T,
                                    float* __restrict__ agg, int e) {
    unsigned gid = blockIdx.x * blockDim.x + threadIdx.x;
    unsigned ei  = gid >> 6;
    unsigned f   = gid & 63u;
    if (ei >= (unsigned)e) return;
    int s = src[ei], d = dst[ei];
    float v = T[(size_t)s * 64 + f] * norm[ei];
    __hip_atomic_fetch_add(&agg[(size_t)d * 64 + f], v,
                           __ATOMIC_RELAXED, __HIP_MEMORY_SCOPE_AGENT);
}

// ---------------------------------------------------------------------------
// Epilogue: h = relu(agg + b); jk = max(jk, h); hb = f16(h)
// ---------------------------------------------------------------------------
__global__ void bias_relu_jk_kernel(const float* __restrict__ agg,
                                    const float* __restrict__ bias,
                                    float* __restrict__ jk,
                                    _Float16* __restrict__ hb,
                                    int total, int first) {
    int i = blockIdx.x * blockDim.x + threadIdx.x;
    if (i >= total) return;
    float h = agg[i] + bias[i & 63];
    h = fmaxf(h, 0.0f);
    jk[i] = first ? h : fmaxf(jk[i], h);
    hb[i] = (_Float16)h;
}

// ---------------------------------------------------------------------------
// FC (64x40) + log_softmax, fp32, LDS-tiled. 64 threads/block, 1 node/thread.
// ---------------------------------------------------------------------------
__global__ void fc_logsoftmax_kernel(const float* __restrict__ jk,
                                     const float* __restrict__ fcw,
                                     const float* __restrict__ fcb,
                                     float* __restrict__ out, int n) {
    __shared__ float sW[HDIM * NCLS];
    __shared__ float sB[NCLS];
    __shared__ float sJ[64][HDIM + 1];              // +1 pad: no bank conflicts

    const int t = threadIdx.x;                      // blockDim.x == 64
    for (int i = t; i < HDIM * NCLS; i += 64) sW[i] = fcw[i];
    if (t < NCLS) sB[t] = fcb[t];

    const int node0 = blockIdx.x * 64;
    for (int r = 0; r < 64; ++r) {
        int nd = node0 + r;
        sJ[r][t] = (nd < n) ? jk[(size_t)nd * HDIM + t] : 0.0f;
    }
    __syncthreads();

    const int node = node0 + t;
    if (node >= n) return;

    float logit[NCLS];
    float mx = -3.0e38f;
#pragma unroll 4
    for (int c = 0; c < NCLS; ++c) {
        float s = sB[c];
        for (int k = 0; k < HDIM; ++k) s += sJ[t][k] * sW[k * NCLS + c];
        logit[c] = s;
        mx = fmaxf(mx, s);
    }
    float sum = 0.0f;
    for (int c = 0; c < NCLS; ++c) sum += __expf(logit[c] - mx);
    float lse = mx + __logf(sum);
    float* orow = out + (size_t)node * NCLS;
    for (int c = 0; c < NCLS; ++c) orow[c] = logit[c] - lse;
}

// ---------------------------------------------------------------------------
// Host orchestration
// ---------------------------------------------------------------------------
static inline size_t align256(size_t x) { return (x + 255) & ~(size_t)255; }

extern "C" void kernel_launch(void* const* d_in, const int* in_sizes, int n_in,
                              void* d_out, int out_size, void* d_ws, size_t ws_size,
                              hipStream_t stream) {
    (void)in_sizes; (void)n_in; (void)out_size; (void)ws_size;

    const float* x    = (const float*)d_in[0];        // [N, 128]
    const int*   eidx = (const int*)  d_in[1];        // [2, E]
    const float* W0   = (const float*)d_in[2];        // [128, 64]
    const float* Ws   = (const float*)d_in[3];        // [5, 64, 64]
    const float* bs   = (const float*)d_in[4];        // [6, 64]
    const float* fcw  = (const float*)d_in[5];        // [64, 40]
    const float* fcb  = (const float*)d_in[6];        // [40]
    float*       out  = (float*)d_out;                // [N, 40]

    const int* src = eidx;                            // edge_index[0]
    const int* dst = eidx + N_EDGES;                  // edge_index[1]

    // ---- workspace carve-up -------------------------------------------------
    char* ws = (char*)d_ws;
    size_t off = 0;
    float*    dis  = (float*)   (ws + off); off += align256((size_t)N_NODES * 4);
    float*    nrm  = (float*)   (ws + off); off += align256((size_t)N_EDGES * 4);
    _Float16* xb   = (_Float16*)(ws + off); off += align256((size_t)N_NODES * F_IN * 2);
    _Float16* hb   = (_Float16*)(ws + off); off += align256((size_t)N_NODES * HDIM * 2);
    float*    tbuf = (float*)   (ws + off); off += align256((size_t)N_NODES * HDIM * 4);
    float*    agg  = (float*)   (ws + off); off += align256((size_t)N_NODES * HDIM * 4);
    float*    jk   = (float*)   (ws + off); off += align256((size_t)N_NODES * HDIM * 4);
    _Float16* W0p  = (_Float16*)(ws + off); off += align256((size_t)F_IN * HDIM * 2);
    _Float16* Wsp  = (_Float16*)(ws + off); off += align256((size_t)5 * HDIM * HDIM * 2);

    const int B = 256;

    // ---- GCN normalization --------------------------------------------------
    deg_init_kernel <<<(N_NODES + B - 1) / B, B, 0, stream>>>(dis, N_NODES);
    deg_count_kernel<<<(N_EDGES + B - 1) / B, B, 0, stream>>>(dst, dis, N_EDGES);
    deg_rsqrt_kernel<<<(N_NODES + B - 1) / B, B, 0, stream>>>(dis, N_NODES);
    edge_norm_kernel<<<(N_EDGES + B - 1) / B, B, 0, stream>>>(src, dst, dis, nrm, N_EDGES);

    // ---- one-time conversions / weight packing ------------------------------
    cvt_f16_kernel<<<((N_NODES * F_IN) + B - 1) / B, B, 0, stream>>>(
        x, xb, N_NODES * F_IN);
    pack_w_kernel<<<((F_IN * HDIM) + B - 1) / B, B, 0, stream>>>(W0, W0p, F_IN);
    for (int l = 0; l < 5; ++l)
        pack_w_kernel<<<((HDIM * HDIM) + B - 1) / B, B, 0, stream>>>(
            Ws + (size_t)l * HDIM * HDIM, Wsp + (size_t)l * HDIM * HDIM, HDIM);

    // ---- 6 GCN layers -------------------------------------------------------
    const int totalNH   = N_NODES * HDIM;
    const int gemmGrid  = ((N_NODES / 16) + 7) / 8;           // 8 waves/block
    const unsigned scatterGrid = ((unsigned)N_EDGES * 64u + B - 1) / B;

    for (int l = 0; l < NLAYERS; ++l) {
        const _Float16* A  = (l == 0) ? xb : hb;
        const int       K  = (l == 0) ? F_IN : HDIM;
        const _Float16* Bp = (l == 0) ? W0p : (Wsp + (size_t)(l - 1) * HDIM * HDIM);

        gemm_wmma_kernel<<<gemmGrid, B, 0, stream>>>(A, Bp, tbuf, N_NODES, K);
        self_init_kernel<<<(totalNH + B - 1) / B, B, 0, stream>>>(
            tbuf, dis, agg, totalNH);
        edge_scatter_kernel<<<scatterGrid, B, 0, stream>>>(
            src, dst, nrm, tbuf, agg, N_EDGES);
        bias_relu_jk_kernel<<<(totalNH + B - 1) / B, B, 0, stream>>>(
            agg, bs + (size_t)l * HDIM, jk, hb, totalNH, l == 0 ? 1 : 0);
    }

    // ---- FC + log_softmax ---------------------------------------------------
    fc_logsoftmax_kernel<<<(N_NODES + 63) / 64, 64, 0, stream>>>(
        jk, fcw, fcb, out, N_NODES);
}
